// PositionOnlyAtt_9002251453071
// MI455X (gfx1250) — matrix operand order repaired and checked by reading, not compiled, not run on hardware
//
#include <hip/hip_runtime.h>
#include <hip/hip_bf16.h>

#ifndef __has_builtin
#define __has_builtin(x) 0
#endif

#if defined(__AMDGCN__) && __has_builtin(__builtin_amdgcn_wmma_f32_16x16x4_f32)
#define USE_F32_WMMA 1
#endif

#if defined(__AMDGCN__) && __has_builtin(__builtin_amdgcn_tensor_load_to_lds)
#define USE_TDM 1
#endif

typedef __attribute__((ext_vector_type(2)))  float        v2f;
typedef __attribute__((ext_vector_type(8)))  float        v8f;
typedef __attribute__((ext_vector_type(16))) _Float16     v16h;
typedef __attribute__((ext_vector_type(4)))  unsigned int v4u;
typedef __attribute__((ext_vector_type(4)))  int          v4i;
typedef __attribute__((ext_vector_type(8)))  int          v8i;

// Problem constants (reference: CH=128, KSIZE=7, BS=8, H=W=56)
#define CHN   128
#define RNB   49          // 7*7 neighborhood
#define NQ    3136        // 56*56 positions
#define XB    401408      // per-batch elements of x = 128*3136

// ---- analytic relative-coordinate math (closed form of the reference's
//      permute/reshape layout reinterpretation + pad + double-unfold) ----
__device__ __forceinline__ float grid_coord(int t) {
    return -3.0f + 6.0f * (float)t * (1.0f / 55.0f);
}
__device__ __forceinline__ float coord_comp(int p, int a) {
    return (a == 0) ? grid_coord(p / 56) : grid_coord(p % 56);
}
__device__ __forceinline__ float ycoord(int u, int vp, int ep) {
    int flatp = u * 112 + vp * 2 + ep;          // < 6272
    int a = (flatp >= NQ) ? 1 : 0;
    int p = flatp - a * NQ;
    return coord_comp(p, a);
}
// rel[q, r, c2] = patches[q, r, c2] - coords[q, c2]
__device__ __forceinline__ float rel_val(int q, int r, int c2) {
    int idx = q * 98 + c2 * 49 + r;             // flat index into (7,7,56,56,2)
    int i  = idx / 43904;                       // 7*56*56*2
    int t  = idx - i * 43904;
    int j  = t / 6272;                          // 56*56*2
    t     -= j * 6272;
    int u  = t / 112;
    t     -= u * 112;
    int v  = t >> 1;
    int e  = t & 1;
    int vv = v + i - 3;
    int ee = e + j - 3;
    float pv = 0.0f;
    if (vv >= 0 && vv < 56 && ee >= 0 && ee < 2) pv = ycoord(u, vv, ee);
    return pv - coord_comp(q, c2);
}
// layer-1 hidden activation H[m,k] = relu(rel0*W1[0,k] + rel1*W1[1,k] + b1[k])
__device__ __forceinline__ float hidden_val(float r0, float r1,
                                            const float* __restrict__ W1,
                                            const float* __restrict__ b1, int k) {
    float h = fmaf(r0, W1[k], fmaf(r1, W1[CHN + k], b1[k]));
    return h > 0.0f ? h : 0.0f;
}

// Fused: [MLP layer1 (VALU) -> layer2 GEMM (fp32 WMMA, B from TDM-staged LDS)
//        -> softmax(r) in LDS -> per-channel weighted gather-reduce]
// One workgroup per position q; 256 threads = 8 wave32.
// Dynamic LDS: [0, 64KB) = W2 tile, [64KB, 96KB) = padded 49x128 logits.
__global__ __launch_bounds__(256) void pos_att_fused(
    const float* __restrict__ x,   // (8,128,56,56)
    const float* __restrict__ W1,  // (2,128)
    const float* __restrict__ b1,  // (128,)
    const float* __restrict__ W2,  // (128,128)
    const float* __restrict__ b2,  // (128,)
    float* __restrict__ out)       // (8,128,56,56)
{
    extern __shared__ float smem[];
    float* sW2 = smem;                  // 128*128 floats = 64 KB
    float* sP  = smem + CHN * CHN;      // 64*128 floats  = 32 KB

    const int q    = blockIdx.x;
    const int tid  = threadIdx.x;
    const int lane = tid & 31;
    const int wave = tid >> 5;

    // Warm the x tile for phase 3 while the TDM stages W2.
    if (tid < 64) __builtin_prefetch(x + (q * 128 + tid * 64) % XB, 0, 0);

    // ------------- Stage W2 (128x128 fp32) into LDS via Tensor Data Mover ----
#ifdef USE_TDM
    if (wave == 0) {
        // D# group 0: count=1 (valid, user), lds_addr, 57b global_addr, type=2
        unsigned long long ga = (unsigned long long)(size_t)W2;
        unsigned lds_off = (unsigned)(size_t)sW2;       // LDS byte offset (low 32b)
        v4u g0;
        g0[0] = 1u;
        g0[1] = lds_off;
        g0[2] = (unsigned)(ga & 0xFFFFFFFFu);
        g0[3] = (unsigned)((ga >> 32) & 0x01FFFFFFu) | 0x80000000u;
        // D# group 1: data_size=4B; tensor 128x128; tile 128x128; dim0 stride 128
        v8i g1;
        g1[0] = 0x00020000;          // [17:16] data_size = 2 (4 bytes); wg_mask=0
        g1[1] = (int)(128u << 16);   // tensor_dim0[15:0] @ bit48
        g1[2] = (int)(128u << 16);   // tensor_dim1[15:0] @ bit80
        g1[3] = (int)(128u << 16);   // tile_dim0 @ bit112
        g1[4] = 128;                 // tile_dim1 @ bit128
        g1[5] = 128;                 // tensor_dim0_stride[31:0] @ bit160
        g1[6] = 0;
        g1[7] = 0;
        v4i z4 = {0, 0, 0, 0};
#if __clang_major__ >= 23
        v8i z8 = {0, 0, 0, 0, 0, 0, 0, 0};
        __builtin_amdgcn_tensor_load_to_lds(g0, g1, z4, z4, z8, 0);
#else
        __builtin_amdgcn_tensor_load_to_lds(g0, g1, z4, z4, 0);
#endif
#if __has_builtin(__builtin_amdgcn_s_wait_tensorcnt)
        __builtin_amdgcn_s_wait_tensorcnt((short)0);
#endif
    }
#else
    for (int t = tid; t < CHN * CHN; t += 256) sW2[t] = W2[t];
#endif
    __syncthreads();

    // ---------------- Phase 1: P = relu(rel@W1+b1)@W2 + b2 via WMMA ----------
    const int c0   = wave * 16;                // this wave's 16-column block
    const int nlo  = lane & 15;
    const int hi   = lane >> 4;                // half-wave selector
    const int bcol = c0 + nlo;                 // B/D column for this lane
    const float bias = b2[bcol];

    for (int mb = 0; mb < 4; ++mb) {           // 4 row tiles cover 49 (pad 64) rows
        const int lr = mb * 16 + nlo;          // A-row for this lane (fixed over K)
        const float r0 = rel_val(q, lr, 0);    // rows >=49 are padding; harmless
        const float r1 = rel_val(q, lr, 1);
        v8f acc = {};
#ifdef USE_F32_WMMA
        // A 16x4 f32: lane<16 -> K = k0+{0,1}; lane>=16 -> K = k0+{2,3}
        const int kb = hi << 1;
        for (int k0 = 0; k0 < CHN; k0 += 4) {
            v2f a, b;
            a[0] = hidden_val(r0, r1, W1, b1, k0 + kb);
            a[1] = hidden_val(r0, r1, W1, b1, k0 + kb + 1);
            b[0] = sW2[(k0 + kb) * CHN + bcol];
            b[1] = sW2[(k0 + kb + 1) * CHN + bcol];
            acc = __builtin_amdgcn_wmma_f32_16x16x4_f32(
                false, a, false, b, (short)0, acc, false, false);
        }
#else
        // Fallback: f16 16x16x32 (codegen-confirmed builtin)
        for (int k0 = 0; k0 < CHN; k0 += 32) {
            v16h a, b;
            #pragma unroll
            for (int g = 0; g < 16; ++g) {
                int ka = k0 + (g & 7) + ((g >> 3) << 4) + (hi << 3);
                a[g] = (_Float16)hidden_val(r0, r1, W1, b1, ka);
                int kbb = k0 + (g & 7) + ((g >> 3) << 4) + (hi << 3);
                b[g] = (_Float16)sW2[kbb * CHN + bcol];
            }
            acc = __builtin_amdgcn_wmma_f32_16x16x32_f16(
                false, a, false, b, (short)0, acc, false, false);
        }
#endif
        // C/D layout: VGPR g, lane -> row = g + 8*hi, col = nlo
        #pragma unroll
        for (int g = 0; g < 8; ++g)
            sP[(mb * 16 + g + 8 * hi) * CHN + bcol] = acc[g] + bias;
    }
    __syncthreads();

    // ---------------- Phase 2: softmax over r (rows 0..48) per column --------
    if (tid < CHN) {
        const int c = tid;
        float mx = sP[c];
        for (int r = 1; r < RNB; ++r) mx = fmaxf(mx, sP[r * CHN + c]);
        float s = 0.0f;
        for (int r = 0; r < RNB; ++r) {
            float e = __expf(sP[r * CHN + c] - mx);
            sP[r * CHN + c] = e;
            s += e;
        }
        float inv = 1.0f / s;
        for (int r = 0; r < RNB; ++r) sP[r * CHN + c] *= inv;
    }
    __syncthreads();

    // ---------------- Phase 3: out[b,q,c] = sum_r Wsm[r,c] * V[b,q,r,c] ------
    // thread -> channel c = tid&127, batches {bsel, bsel+2, bsel+4, bsel+6}
    const int c    = tid & (CHN - 1);
    const int bsel = tid >> 7;
    float accv[4] = {0.f, 0.f, 0.f, 0.f};
    for (int r = 0; r < RNB; ++r) {
        // invert the reference's (7,7,56,56,128)->(n,c,r) reinterpretation
        int idx = q * 6272 + c * 49 + r;
        int i  = idx / 2809856;                // 7*56*56*128
        int t2 = idx - i * 2809856;
        int j  = t2 / XB;                      // 56*56*128
        t2    -= j * XB;
        int u  = t2 / 7168;                    // 56*128
        t2    -= u * 7168;
        int v  = t2 >> 7;
        int e  = t2 & 127;
        int vv = v + i - 3;
        int ee = e + j - 3;
        if (vv >= 0 && vv < 56 && ee >= 0 && ee < CHN) {
            const float w  = sP[r * CHN + c];
            const int base = u * 7168 + vv * CHN + ee;   // x is y reinterpreted
            #pragma unroll
            for (int bb = 0; bb < 4; ++bb) {
                int b = bsel + (bb << 1);
                accv[bb] = fmaf(w, x[b * XB + base], accv[bb]);
            }
        }
    }
    #pragma unroll
    for (int bb = 0; bb < 4; ++bb) {
        int b = bsel + (bb << 1);
        out[b * XB + c * NQ + q] = accv[bb];
    }
}

extern "C" void kernel_launch(void* const* d_in, const int* in_sizes, int n_in,
                              void* d_out, int out_size, void* d_ws, size_t ws_size,
                              hipStream_t stream) {
    (void)in_sizes; (void)n_in; (void)d_ws; (void)ws_size; (void)out_size;
    const float* x  = (const float*)d_in[0];
    const float* W1 = (const float*)d_in[1];
    const float* b1 = (const float*)d_in[2];
    const float* W2 = (const float*)d_in[3];
    const float* b2 = (const float*)d_in[4];
    // d_in[5] is ksize (==7), baked in at compile time
    float* out = (float*)d_out;
    const size_t lds_bytes = (size_t)(CHN * CHN + 64 * CHN) * sizeof(float); // 96 KB
    pos_att_fused<<<dim3(NQ), dim3(256), lds_bytes, stream>>>(x, W1, b1, W2, b2, out);
}
